// CycleBlock_1382979469472
// MI455X (gfx1250) — matrix operand order repaired and checked by reading, not compiled, not run on hardware
//
#include <hip/hip_runtime.h>
#include <hip/hip_bf16.h>

// ---------------------------------------------------------------------------
// CycleBlock on MI455X (gfx1250): bf16 WMMA GEMMs with f32 accumulation.
// B=16, H=56, W=56, C=384, tokens = 50176 (3136 per batch, 196 M-tiles each).
// ---------------------------------------------------------------------------

typedef __attribute__((ext_vector_type(16))) __bf16 v16bf;
typedef __attribute__((ext_vector_type(8)))  __bf16 v8bf;
typedef __attribute__((ext_vector_type(8)))  float  v8f;

#define CB_B    16
#define CB_H    56
#define CB_W    56
#define CB_C    384
#define CB_TOK  (CB_B * CB_H * CB_W)   // 50176
#define CB_TOKB (CB_H * CB_W)          // 3136 tokens per batch (196 tiles of 16)
#define CB_NT   6                      // 384 / 64 N-tiles per wave row
#define CB_WAVES (CB_TOK / 16 * CB_NT) // 18816 wave-tiles
#define CB_BLOCKS (CB_WAVES / 8)       // 2352 blocks of 8 waves

__device__ __forceinline__ unsigned short f2bf(float f) {
    unsigned int u = __builtin_bit_cast(unsigned int, f);
    u += 0x7FFFu + ((u >> 16) & 1u);          // round-to-nearest-even
    return (unsigned short)(u >> 16);
}
__device__ __forceinline__ float bfu2f(unsigned int lo16) {
    unsigned int u = lo16 << 16;
    return __builtin_bit_cast(float, u);
}

// ---------------------------------------------------------------------------
// 1) Convert + transpose the three 384x384 weights: wT[n][k] = bf16(w[k][n]).
// ---------------------------------------------------------------------------
__global__ void k_convert_weights(const float* __restrict__ wh,
                                  const float* __restrict__ wc,
                                  const float* __restrict__ wp,
                                  unsigned short* __restrict__ whT,
                                  unsigned short* __restrict__ wcT,
                                  unsigned short* __restrict__ wpT) {
    int idx = blockIdx.x * blockDim.x + threadIdx.x;
    const int per = CB_C * CB_C;
    if (idx >= 3 * per) return;
    int which = idx / per;
    int r     = idx - which * per;     // r = k*384 + n
    int k = r / CB_C, n = r - k * CB_C;
    const float* src = (which == 0) ? wh : (which == 1) ? wc : wp;
    unsigned short* dst = (which == 0) ? whT : (which == 1) ? wcT : wpT;
    dst[n * CB_C + k] = f2bf(src[r]);
}

// ---------------------------------------------------------------------------
// 2) x -> bf16, and per-channel vertically shifted xs -> bf16 (zero fill).
//    shift pattern period 8: [-2,-1,0,1,2,1,0,-1]
// ---------------------------------------------------------------------------
__global__ void k_convert_x(const float* __restrict__ x,
                            unsigned short* __restrict__ xbf,
                            unsigned short* __restrict__ xsbf) {
    int idx = blockIdx.x * blockDim.x + threadIdx.x;
    if (idx >= CB_TOK * CB_C) return;
    int t = idx / CB_C;
    int c = idx - t * CB_C;
    int b  = t / CB_TOKB;
    int hw = t - b * CB_TOKB;
    int h = hw / CB_W;
    int w = hw - h * CB_W;

    xbf[idx] = f2bf(x[idx]);

    const int SH[8] = {-2, -1, 0, 1, 2, 1, 0, -1};
    int h2 = h - SH[c & 7];
    float v = 0.0f;
    if (h2 >= 0 && h2 < CB_H)
        v = x[(((size_t)b * CB_H + h2) * CB_W + w) * CB_C + c];
    xsbf[idx] = f2bf(v);
}

// ---------------------------------------------------------------------------
// 3) Wave-tiled bf16 GEMM: Out[50176x384] = bf16(A[50176x384] @ W + bias).
//    Each wave: 16M x 64N tile, K=384 in 12 steps of 32.
//    Also accumulates coef * column-sums into accum[b][c] (for the mean).
// ---------------------------------------------------------------------------
__global__ __launch_bounds__(256)
void k_gemm_bias_acc(const unsigned short* __restrict__ A,
                     const unsigned short* __restrict__ Wt,   // [n][k] bf16
                     const float* __restrict__ bias,
                     unsigned short* __restrict__ Out,
                     float* __restrict__ accum,
                     float coef) {
    int wave = blockIdx.x * (blockDim.x >> 5) + (threadIdx.x >> 5);
    if (wave >= CB_WAVES) return;                 // wave-uniform
    int tm = wave / CB_NT;
    int tn = wave - tm * CB_NT;
    int lane = threadIdx.x & 31;
    int mrow = lane & 15;
    int hi   = lane >> 4;
    int row0 = tm << 4;
    int n0   = tn << 6;
    int b    = row0 / CB_TOKB;

    v8f acc[4];
#pragma unroll
    for (int t = 0; t < 4; ++t) {
        float bb = bias[n0 + t * 16 + mrow];
#pragma unroll
        for (int i = 0; i < 8; ++i) acc[t][i] = bb;
    }

    const unsigned short* arow = A + (size_t)(row0 + mrow) * CB_C;
    for (int kk = 0; kk < CB_C; kk += 32) {
        // A fragment: lane<16 -> K {kk..kk+15}; lane>=16 -> +16
        v8bf c0 = *(const v8bf*)(arow + kk + hi * 16);
        v8bf c1 = *(const v8bf*)(arow + kk + 8 + hi * 16);
        v16bf afrag;
#pragma unroll
        for (int i = 0; i < 8; ++i) { afrag[i] = c0[i]; afrag[i + 8] = c1[i]; }
#pragma unroll
        for (int t = 0; t < 4; ++t) {
            const unsigned short* bp =
                Wt + (size_t)(n0 + t * 16 + mrow) * CB_C + kk + hi * 16;
            v16bf bfrag = *(const v16bf*)bp;
            acc[t] = __builtin_amdgcn_wmma_f32_16x16x32_bf16(
                false, afrag, false, bfrag, (short)0, acc[t], false, false);
        }
    }

#pragma unroll
    for (int t = 0; t < 4; ++t) {
        int col = n0 + t * 16 + mrow;
        float csum = 0.0f;
#pragma unroll
        for (int v = 0; v < 8; ++v) {
            float val = acc[t][v];
            Out[(size_t)(row0 + v + hi * 8) * CB_C + col] = f2bf(val);
            csum += val;
        }
        atomicAdd(&accum[b * CB_C + col], coef * csum);
    }
}

// ---------------------------------------------------------------------------
// 4) Reweight MLP: mean -> GELU(a@w_r1+b_r1)@w_r2+b_r2 -> softmax over 3.
//    Writes packed bf16 scale pairs: lo = p0+p1 (for x_h), hi = p2 (for x_c).
// ---------------------------------------------------------------------------
__global__ void k_reweight(const float* __restrict__ accum,
                           const float* __restrict__ wr1, const float* __restrict__ br1,
                           const float* __restrict__ wr2, const float* __restrict__ br2,
                           unsigned int* __restrict__ scpk) {
    __shared__ float aa[CB_C];
    __shared__ float hid[96];
    __shared__ float zz[1152];
    int b = blockIdx.x, tid = threadIdx.x, nthr = blockDim.x;

    for (int c = tid; c < CB_C; c += nthr)
        aa[c] = accum[b * CB_C + c] * (1.0f / (float)CB_TOKB);
    __syncthreads();

    for (int j = tid; j < 96; j += nthr) {
        float s = br1[j];
        for (int c = 0; c < CB_C; ++c) s += aa[c] * wr1[c * 96 + j];
        hid[j] = 0.5f * s * (1.0f + erff(s * 0.70710678118654752f));  // exact GELU
    }
    __syncthreads();

    for (int o = tid; o < 1152; o += nthr) {
        float s = br2[o];
        for (int j = 0; j < 96; ++j) s += hid[j] * wr2[j * 1152 + o];
        zz[o] = s;
    }
    __syncthreads();

    for (int c = tid; c < CB_C; c += nthr) {
        float l0 = zz[c * 3 + 0], l1 = zz[c * 3 + 1], l2 = zz[c * 3 + 2];
        float mx = fmaxf(l0, fmaxf(l1, l2));
        float e0 = expf(l0 - mx), e1 = expf(l1 - mx), e2 = expf(l2 - mx);
        float inv = 1.0f / (e0 + e1 + e2);
        float shw = (e0 + e1) * inv;   // multiplies x_h (== x_w term folded in)
        float scv = e2 * inv;          // multiplies x_c
        scpk[b * CB_C + c] =
            (unsigned)f2bf(shw) | ((unsigned)f2bf(scv) << 16);
    }
}

// ---------------------------------------------------------------------------
// 5) Final GEMM: out = (s_hw*x_h + s_c*x_c) @ w_p + b_p, f32 output.
//    Scale-combine fused into the A-fragment build.
// ---------------------------------------------------------------------------
__global__ __launch_bounds__(256)
void k_gemm_final(const unsigned short* __restrict__ xh,
                  const unsigned short* __restrict__ xc,
                  const unsigned int* __restrict__ scpk,
                  const unsigned short* __restrict__ Wt,
                  const float* __restrict__ bias,
                  float* __restrict__ out) {
    int wave = blockIdx.x * (blockDim.x >> 5) + (threadIdx.x >> 5);
    if (wave >= CB_WAVES) return;
    int tm = wave / CB_NT;
    int tn = wave - tm * CB_NT;
    int lane = threadIdx.x & 31;
    int mrow = lane & 15;
    int hi   = lane >> 4;
    int row0 = tm << 4;
    int n0   = tn << 6;
    int b    = row0 / CB_TOKB;

    v8f acc[4];
#pragma unroll
    for (int t = 0; t < 4; ++t) {
        float bb = bias[n0 + t * 16 + mrow];
#pragma unroll
        for (int i = 0; i < 8; ++i) acc[t][i] = bb;
    }

    const unsigned short* hrow = xh + (size_t)(row0 + mrow) * CB_C;
    const unsigned short* crow = xc + (size_t)(row0 + mrow) * CB_C;
    const unsigned int*   sp   = scpk + b * CB_C;

    for (int kk = 0; kk < CB_C; kk += 32) {
        int c0b = kk + hi * 16;
        int c1b = kk + 8 + hi * 16;
        v8bf h0 = *(const v8bf*)(hrow + c0b);
        v8bf h1 = *(const v8bf*)(hrow + c1b);
        v8bf g0 = *(const v8bf*)(crow + c0b);
        v8bf g1 = *(const v8bf*)(crow + c1b);
        v16bf afrag;
#pragma unroll
        for (int i = 0; i < 8; ++i) {
            unsigned p0 = sp[c0b + i];
            afrag[i] = (__bf16)(bfu2f(p0 & 0xFFFFu) * (float)h0[i] +
                                bfu2f(p0 >> 16)     * (float)g0[i]);
            unsigned p1 = sp[c1b + i];
            afrag[i + 8] = (__bf16)(bfu2f(p1 & 0xFFFFu) * (float)h1[i] +
                                    bfu2f(p1 >> 16)     * (float)g1[i]);
        }
#pragma unroll
        for (int t = 0; t < 4; ++t) {
            const unsigned short* bp =
                Wt + (size_t)(n0 + t * 16 + mrow) * CB_C + kk + hi * 16;
            v16bf bfrag = *(const v16bf*)bp;
            acc[t] = __builtin_amdgcn_wmma_f32_16x16x32_bf16(
                false, afrag, false, bfrag, (short)0, acc[t], false, false);
        }
    }

#pragma unroll
    for (int t = 0; t < 4; ++t) {
        int col = n0 + t * 16 + mrow;
#pragma unroll
        for (int v = 0; v < 8; ++v)
            out[(size_t)(row0 + v + hi * 8) * CB_C + col] = acc[t][v];
    }
}

// ---------------------------------------------------------------------------
// Host launcher. Workspace layout (256B aligned regions), ~148 MB total:
//   whT/wcT/wpT bf16 (3 x 294912 B), accum f32 (24576 B), scpk (24576 B),
//   x_bf16 / xs_bf16 / xh_bf16 / xc_bf16 (4 x 38535168 B).
// ---------------------------------------------------------------------------
extern "C" void kernel_launch(void* const* d_in, const int* in_sizes, int n_in,
                              void* d_out, int out_size, void* d_ws, size_t ws_size,
                              hipStream_t stream) {
    (void)in_sizes; (void)n_in; (void)out_size; (void)ws_size;
    const float* x    = (const float*)d_in[0];
    const float* w_h  = (const float*)d_in[1];
    const float* b_h  = (const float*)d_in[2];
    const float* w_c  = (const float*)d_in[3];
    const float* b_c  = (const float*)d_in[4];
    const float* w_r1 = (const float*)d_in[5];
    const float* b_r1 = (const float*)d_in[6];
    const float* w_r2 = (const float*)d_in[7];
    const float* b_r2 = (const float*)d_in[8];
    const float* w_p  = (const float*)d_in[9];
    const float* b_p  = (const float*)d_in[10];

    char* ws = (char*)d_ws;
    const size_t WSZ  = (size_t)CB_C * CB_C * 2;           // 294912
    const size_t XSZ  = (size_t)CB_TOK * CB_C * 2;         // 38535168
    unsigned short* whT  = (unsigned short*)(ws);
    unsigned short* wcT  = (unsigned short*)(ws + WSZ);
    unsigned short* wpT  = (unsigned short*)(ws + 2 * WSZ);
    float*          accum= (float*)(ws + 3 * WSZ);                        // 24576 B
    unsigned int*   scpk = (unsigned int*)(ws + 3 * WSZ + 24576);         // 24576 B
    unsigned short* x_bf = (unsigned short*)(ws + 3 * WSZ + 49152);
    unsigned short* xs_bf= (unsigned short*)(ws + 3 * WSZ + 49152 + XSZ);
    unsigned short* xh_bf= (unsigned short*)(ws + 3 * WSZ + 49152 + 2 * XSZ);
    unsigned short* xc_bf= (unsigned short*)(ws + 3 * WSZ + 49152 + 3 * XSZ);

    hipMemsetAsync(accum, 0, (size_t)CB_B * CB_C * sizeof(float), stream);

    k_convert_weights<<<(3 * CB_C * CB_C + 255) / 256, 256, 0, stream>>>(
        w_h, w_c, w_p, whT, wcT, wpT);

    k_convert_x<<<(CB_TOK * CB_C) / 256, 256, 0, stream>>>(x, x_bf, xs_bf);

    // x_h = xs @ w_h + b_h ; contributes 2x to the mean (x_h + x_w)
    k_gemm_bias_acc<<<CB_BLOCKS, 256, 0, stream>>>(xs_bf, whT, b_h, xh_bf, accum, 2.0f);
    // x_c = x @ w_c + b_c ; contributes 1x
    k_gemm_bias_acc<<<CB_BLOCKS, 256, 0, stream>>>(x_bf, wcT, b_c, xc_bf, accum, 1.0f);

    k_reweight<<<CB_B, 128, 0, stream>>>(accum, w_r1, b_r1, w_r2, b_r2, scpk);

    k_gemm_final<<<CB_BLOCKS, 256, 0, stream>>>(xh_bf, xc_bf, scpk, wpT, b_p,
                                                (float*)d_out);
}
